// mdLSTM_7722351198776
// MI455X (gfx1250) — matrix operand order repaired and checked by reading, not compiled, not run on hardware
//
#include <hip/hip_runtime.h>
#include <hip/hip_bf16.h>

typedef _Float16 v16h __attribute__((ext_vector_type(16)));
typedef _Float16 v8h  __attribute__((ext_vector_type(8)));
typedef float    v8f  __attribute__((ext_vector_type(8)));

// ---------- constants ----------
#define S_CELLS 4096
#define T_STEPS 64
#define I_FEAT  64
#define EH_DIM  128
#define E_DIM   256
#define H_DIM   512
#define K_TOT   768   // E + H
#define MT      262144 // S*T

// ---------- math helpers ----------
__device__ __forceinline__ float sigmoidf_(float x) {
    return 1.0f / (1.0f + __expf(-x));
}
__device__ __forceinline__ float tanhf_(float x) {
    x = fminf(fmaxf(x, -15.0f), 15.0f);
    float e = __expf(2.0f * x);
    return (e - 1.0f) / (e + 1.0f);
}

// ---------- CDNA5 async global->LDS DMA (ASYNCcnt-tracked) ----------
__device__ __forceinline__ void async_ld16(void* ldsDst, const void* gsrc) {
    unsigned d = (unsigned)(uintptr_t)ldsDst;                 // low 32 bits = LDS offset
    unsigned long long g = (unsigned long long)(uintptr_t)gsrc;
    asm volatile("global_load_async_to_lds_b128 %0, %1, off"
                 :: "v"(d), "v"(g) : "memory");
}
template<int N>
__device__ __forceinline__ void wait_async() {
    asm volatile("s_wait_asynccnt %0" :: "n"(N) : "memory");
}

// ---------- WMMA fragment loaders (16-bit 16x32 A, 32x16 B per CDNA5 ISA) ----------
// A (16x32, MxK): lane<16 -> row=lane, K 0..7 & 16..23 ; lane>=16 -> row=lane-16, K 8..15 & 24..31
__device__ __forceinline__ v16h frag_a(const _Float16* lds, int ldl, int row16, int lane) {
    int r  = row16 + (lane & 15);
    int c0 = (lane >> 4) * 8;
    const v8h lo = *(const v8h*)(lds + r * ldl + c0);
    const v8h hi = *(const v8h*)(lds + r * ldl + c0 + 16);
    return __builtin_shufflevector(lo, hi, 0,1,2,3,4,5,6,7,8,9,10,11,12,13,14,15);
}
// B (32x16, KxN): lane<16 -> col=lane, K 0..7 & 16..23 ; lane>=16 -> col=lane-16, K 8..15 & 24..31
__device__ __forceinline__ v16h frag_b(const _Float16* lds, int ldl, int n0, int lane) {
    const int n  = n0 + (lane & 15);
    const int kh = (lane >> 4) * 8;
    v16h b;
#pragma unroll
    for (int s = 0; s < 8; ++s) {
        b[s]     = lds[(kh + s) * ldl + n];
        b[8 + s] = lds[(16 + kh + s) * ldl + n];
    }
    return b;
}
__device__ __forceinline__ v8f wmma_acc(v16h a, v16h b, v8f c) {
    return __builtin_amdgcn_wmma_f32_16x16x32_f16(false, a, false, b, (short)0, c, false, false);
}

// ---------- tiny prep kernels ----------
__global__ void k_lane(const float* lane, const float* Wlg1, const float* blg1,
                       const float* Wlg2, const float* blg2, float* lc) {
    float x = lane[0];
    float acc = blg2[0];
    for (int j = 0; j < 32; ++j) {
        float lg = fmaxf(x * Wlg1[j] + blg1[j], 0.0f);
        acc += Wlg2[j] * lg;
    }
    float c = 1.0f / (1.0f + __expf(-acc));
    lc[0] = c;
    lc[1] = 1.0f / c;
}

__global__ void k_wcat(const float* __restrict__ Wih, const float* __restrict__ Whh,
                       _Float16* __restrict__ WcatT) {
    int id = blockIdx.x * 256 + threadIdx.x;       // 768*2048
    int k = id >> 11, n = id & 2047;
    float v = (k < E_DIM) ? Wih[n * E_DIM + k] : Whh[n * H_DIM + (k - E_DIM)];
    WcatT[id] = (_Float16)v;
}

__global__ void k_small_prep(const float* __restrict__ bih, const float* __restrict__ bhh,
                             float* __restrict__ bcat,
                             const float* __restrict__ We1, _Float16* __restrict__ We1T,
                             const float* __restrict__ We2, _Float16* __restrict__ We2T) {
    int id = blockIdx.x * 256 + threadIdx.x;       // 2048 + 8192 + 32768 = 43008
    if (id < 2048) {
        bcat[id] = bih[id] + bhh[id];
    } else if (id < 2048 + 64 * 128) {
        int j = id - 2048;
        int k = j >> 7, n = j & 127;
        We1T[j] = (_Float16)We1[n * 64 + k];
    } else if (id < 2048 + 8192 + 128 * 256) {
        int j = id - 2048 - 8192;
        int k = j >> 8, n = j & 255;
        We2T[j] = (_Float16)We2[n * 128 + k];
    }
}

__global__ void k_zero_state(float* __restrict__ cbuf, _Float16* __restrict__ hpool) {
    int id = blockIdx.x * 256 + threadIdx.x;       // 2097152
    cbuf[id] = 0.0f;
    hpool[id] = (_Float16)0.0f;
}

// ---------- generic WMMA GEMM: C[f16] = relu(A @ B + bias) ----------
// BM=128, BN=128, BK=32. 8 waves, each 32x64.
// AMODE 0: A is f16 row-major. AMODE 1: A is f32 row-major, scaled by *scaleP on load.
template<int AMODE>
__global__ __launch_bounds__(256) void wgemm_k(
    const void* __restrict__ Araw, int lda,
    const _Float16* __restrict__ B, int ldb,
    const float* __restrict__ bias,
    _Float16* __restrict__ C, int ldc,
    int K, const float* __restrict__ scaleP)
{
    constexpr int ALD = 40;   // 32 + 8 pad (halfs)
    constexpr int BLD = 136;  // 128 + 8 pad
    __shared__ _Float16 sA[128 * ALD];
    __shared__ _Float16 sB[32 * BLD];

    const int tid  = threadIdx.x;
    const int wave = tid >> 5, lane = tid & 31;
    const int waveM = wave & 3, waveN = wave >> 2;
    const long mBase = (long)blockIdx.x * 128;
    const long nBase = (long)blockIdx.y * 128;
    const float scale = (AMODE == 1) ? scaleP[0] : 1.0f;

    v8f acc[2][4];
#pragma unroll
    for (int a = 0; a < 2; ++a)
#pragma unroll
        for (int b = 0; b < 4; ++b) acc[a][b] = (v8f)0.0f;

    for (int kb = 0; kb < K; kb += 32) {
        { // stage A tile 128x32
            int row = tid >> 1, col = (tid & 1) * 16;
            _Float16* dst = sA + row * ALD + col;
            if (AMODE == 0) {
                const _Float16* src = (const _Float16*)Araw + (mBase + row) * lda + kb + col;
                async_ld16(dst, src);
                async_ld16(dst + 8, src + 8);
            } else {
                const float* src = (const float*)Araw + (mBase + row) * lda + kb + col;
#pragma unroll
                for (int j = 0; j < 16; ++j) dst[j] = (_Float16)(src[j] * scale);
            }
        }
        { // stage B tile 32x128 (async DMA to LDS)
            int row = tid >> 3, col = (tid & 7) * 16;
            const _Float16* src = B + (long)(kb + row) * ldb + nBase + col;
            _Float16* dst = sB + row * BLD + col;
            async_ld16(dst, src);
            async_ld16(dst + 8, src + 8);
        }
        wait_async<0>();
        __syncthreads();

        v16h af[2], bf[4];
#pragma unroll
        for (int tm = 0; tm < 2; ++tm) af[tm] = frag_a(sA, ALD, waveM * 32 + tm * 16, lane);
#pragma unroll
        for (int tn = 0; tn < 4; ++tn) bf[tn] = frag_b(sB, BLD, waveN * 64 + tn * 16, lane);
#pragma unroll
        for (int tm = 0; tm < 2; ++tm)
#pragma unroll
            for (int tn = 0; tn < 4; ++tn)
                acc[tm][tn] = wmma_acc(af[tm], bf[tn], acc[tm][tn]);
        __syncthreads();
    }

    // epilogue: bias + relu + f16 store
    const int Ncol = lane & 15;
    const int Mhi  = (lane >> 4) * 8;
#pragma unroll
    for (int tm = 0; tm < 2; ++tm)
#pragma unroll
        for (int tn = 0; tn < 4; ++tn) {
            long nG = nBase + waveN * 64 + tn * 16 + Ncol;
            float bv = bias[nG];
#pragma unroll
            for (int r = 0; r < 8; ++r) {
                float v = acc[tm][tn][r] + bv;
                v = fmaxf(v, 0.0f);
                long mG = mBase + waveM * 32 + tm * 16 + Mhi + r;
                C[mG * ldc + nG] = (_Float16)v;
            }
        }
}

// ---------- fused LSTM step: gates GEMM (all 4 gates) + pointwise update ----------
// Grid: (S/128, H/64). 8 waves; wave = (rowStrip 0..3)*32 rows x (colHalf 0..1)*32 cols x 4 gates.
// Double-buffered LDS tiles filled by async global->LDS DMA.
__global__ __launch_bounds__(256) void lstm_step_k(
    const _Float16* __restrict__ xemb,   // [S*T, 256] f16
    const _Float16* __restrict__ hpool,  // [S, 512] f16 (pooled h from prev step)
    const _Float16* __restrict__ WcatT,  // [768, 2048] f16; col = gate*512 + n
    const float* __restrict__ bcat,      // [2048] = bih + bhh
    float* __restrict__ cbuf,            // [S, 512] f32 (in/out)
    _Float16* __restrict__ hraw,         // [S, 512] f16 out (pre-pool h)
    int tstep)
{
    constexpr int ALD = 40;    // 32+8
    constexpr int BLD = 264;   // 256+8
    constexpr int ASZ = 128 * ALD;
    constexpr int BSZ = 32 * BLD;
    __shared__ _Float16 sA[2 * ASZ];
    __shared__ _Float16 sB[2 * BSZ];

    const int tid = threadIdx.x;
    const int wave = tid >> 5, lane = tid & 31;
    const int rowStrip = wave & 3, colHalf = wave >> 2;
    const long sBase = (long)blockIdx.x * 128;
    const int  nBase = blockIdx.y * 64;

    // per-thread staging coordinates
    const int aRow = tid >> 1, aCol = (tid & 1) * 16;          // A: 128x32, 16 halfs/thread
    const int bRow = tid >> 3, bSeg = tid & 7;                 // B: 32x256, 32 halfs/thread
    const int bGcol = (bSeg >> 1) * H_DIM + nBase + (bSeg & 1) * 32;

    auto stage = [&](int kb, int buf) {
        const _Float16* srcA = (kb < E_DIM)
            ? xemb + ((sBase + aRow) * T_STEPS + tstep) * E_DIM + kb + aCol
            : hpool + (sBase + aRow) * H_DIM + (kb - E_DIM) + aCol;
        _Float16* dA = sA + buf * ASZ + aRow * ALD + aCol;
        async_ld16(dA, srcA);
        async_ld16(dA + 8, srcA + 8);
        const _Float16* srcB = WcatT + (long)(kb + bRow) * 2048 + bGcol;
        _Float16* dB = sB + buf * BSZ + bRow * BLD + bSeg * 32;
#pragma unroll
        for (int j = 0; j < 4; ++j) async_ld16(dB + j * 8, srcB + j * 8);
    };

    v8f acc[2][4][2]; // [rowTile][gate][colTile]
#pragma unroll
    for (int tm = 0; tm < 2; ++tm)
#pragma unroll
        for (int g = 0; g < 4; ++g)
#pragma unroll
            for (int ct = 0; ct < 2; ++ct) acc[tm][g][ct] = (v8f)0.0f;

    constexpr int NK = K_TOT / 32;  // 24
    stage(0, 0);
    for (int ks = 0; ks < NK; ++ks) {
        if (ks + 1 < NK) {
            stage((ks + 1) * 32, (ks + 1) & 1);
            wait_async<6>();       // 6 async ops in flight for next tile; current tile done
        } else {
            wait_async<0>();
        }
        __syncthreads();

        const _Float16* cA = sA + (ks & 1) * ASZ;
        const _Float16* cB = sB + (ks & 1) * BSZ;
        v16h af[2];
#pragma unroll
        for (int tm = 0; tm < 2; ++tm)
            af[tm] = frag_a(cA, ALD, rowStrip * 32 + tm * 16, lane);
#pragma unroll
        for (int g = 0; g < 4; ++g)
#pragma unroll
            for (int ct = 0; ct < 2; ++ct) {
                v16h bf = frag_b(cB, BLD, g * 64 + colHalf * 32 + ct * 16, lane);
#pragma unroll
                for (int tm = 0; tm < 2; ++tm)
                    acc[tm][g][ct] = wmma_acc(af[tm], bf, acc[tm][g][ct]);
            }
        __syncthreads();
    }

    // pointwise: i,f,g,o -> c,h
    const int Ncol = lane & 15;
    const int Mhi  = (lane >> 4) * 8;
#pragma unroll
    for (int tm = 0; tm < 2; ++tm)
#pragma unroll
        for (int ct = 0; ct < 2; ++ct) {
            int nG = nBase + colHalf * 32 + ct * 16 + Ncol;
            float bi  = bcat[nG];
            float bf_ = bcat[512 + nG];
            float bg  = bcat[1024 + nG];
            float bo  = bcat[1536 + nG];
#pragma unroll
            for (int r = 0; r < 8; ++r) {
                long s = sBase + rowStrip * 32 + tm * 16 + Mhi + r;
                long idx = s * H_DIM + nG;
                float iv = sigmoidf_(acc[tm][0][ct][r] + bi);
                float fv = sigmoidf_(acc[tm][1][ct][r] + bf_);
                float gv = tanhf_(acc[tm][2][ct][r] + bg);
                float ov = sigmoidf_(acc[tm][3][ct][r] + bo);
                float cn = fv * cbuf[idx] + iv * gv;
                cbuf[idx] = cn;
                hraw[idx] = (_Float16)(ov * tanhf_(cn));
            }
        }
}

// ---------- spatial maxpool (kernel=3, stride=1, -inf pad) over S axis ----------
__global__ void k_pool(const _Float16* __restrict__ hraw, _Float16* __restrict__ hpool) {
    int id = blockIdx.x * 256 + threadIdx.x;   // 262144 threads, 8 halfs each
    long pos = (long)id * 8;
    int s = id >> 6;                           // 512/8 = 64 vec8 per row
    v8h v = *(const v8h*)(hraw + pos);
    float m[8];
#pragma unroll
    for (int j = 0; j < 8; ++j) m[j] = (float)v[j];
    if (s > 0) {
        v8h a = *(const v8h*)(hraw + pos - H_DIM);
#pragma unroll
        for (int j = 0; j < 8; ++j) m[j] = fmaxf(m[j], (float)a[j]);
    }
    if (s < S_CELLS - 1) {
        v8h b = *(const v8h*)(hraw + pos + H_DIM);
#pragma unroll
        for (int j = 0; j < 8; ++j) m[j] = fmaxf(m[j], (float)b[j]);
    }
    v8h o;
#pragma unroll
    for (int j = 0; j < 8; ++j) o[j] = (_Float16)m[j];
    *(v8h*)(hpool + pos) = o;
}

// ---------- output head ----------
__global__ void k_head1(const _Float16* __restrict__ h, const float* __restrict__ W,
                        const float* __restrict__ b, float* __restrict__ Y) {
    int id = blockIdx.x * 256 + threadIdx.x;   // 4096*128
    int s = id >> 7, o = id & 127;
    const _Float16* hr = h + (long)s * H_DIM;
    const float* wr = W + (long)o * H_DIM;
    float acc = b[o];
    for (int k = 0; k < H_DIM; ++k) acc += (float)hr[k] * wr[k];
    Y[id] = acc;                                // no relu on Wout layer
}

__global__ void k_mlp(const float* __restrict__ X, int K, const float* __restrict__ W,
                      const float* __restrict__ b, float* __restrict__ Y, int N) {
    int id = blockIdx.x * 256 + threadIdx.x;
    int s = id / N, o = id % N;
    const float* xr = X + (long)s * K;
    const float* wr = W + (long)o * K;
    float acc = b[o];
    for (int k = 0; k < K; ++k) acc += xr[k] * wr[k];
    Y[id] = fmaxf(acc, 0.0f);
}

__global__ void k_head4(const float* __restrict__ X, const float* __restrict__ Wf3,
                        const float* __restrict__ bf3, const float* __restrict__ lc,
                        float* __restrict__ out) {
    int s = blockIdx.x * 256 + threadIdx.x;    // 4096
    const float* xr = X + (long)s * 64;
    float acc = bf3[0];
    for (int k = 0; k < 64; ++k) acc += xr[k] * Wf3[k];
    out[s] = acc * lc[1];                      // divide by laneC
}

__global__ void k_final(const _Float16* __restrict__ hpool, const float* __restrict__ cbuf,
                        float* __restrict__ dout) {
    int id = blockIdx.x * 256 + threadIdx.x;   // 2097152
    dout[S_CELLS + id] = (float)hpool[id];
    dout[S_CELLS + S_CELLS * H_DIM + id] = cbuf[id];
}

// ---------- launch ----------
extern "C" void kernel_launch(void* const* d_in, const int* in_sizes, int n_in,
                              void* d_out, int out_size, void* d_ws, size_t ws_size,
                              hipStream_t stream) {
    const float* inputData = (const float*)d_in[0];
    const float* lane  = (const float*)d_in[1];
    const float* Wlg1  = (const float*)d_in[2];
    const float* blg1  = (const float*)d_in[3];
    const float* Wlg2  = (const float*)d_in[4];
    const float* blg2  = (const float*)d_in[5];
    const float* We1   = (const float*)d_in[6];
    const float* be1   = (const float*)d_in[7];
    const float* We2   = (const float*)d_in[8];
    const float* be2   = (const float*)d_in[9];
    const float* Wih   = (const float*)d_in[10];
    const float* bih   = (const float*)d_in[11];
    const float* Whh   = (const float*)d_in[12];
    const float* bhh   = (const float*)d_in[13];
    const float* Wout  = (const float*)d_in[14];
    const float* bout  = (const float*)d_in[15];
    const float* Wf1   = (const float*)d_in[16];
    const float* bf1   = (const float*)d_in[17];
    const float* Wf2   = (const float*)d_in[18];
    const float* bf2   = (const float*)d_in[19];
    const float* Wf3   = (const float*)d_in[20];
    const float* bf3   = (const float*)d_in[21];

    char* ws = (char*)d_ws;
    size_t off = 0;
    auto alloc = [&](size_t bytes) { char* p = ws + off; off += (bytes + 255) & ~(size_t)255; return p; };
    float*    lc    = (float*)alloc(8);
    float*    bcat  = (float*)alloc(2048 * 4);
    _Float16* We1T  = (_Float16*)alloc(64 * 128 * 2);
    _Float16* We2T  = (_Float16*)alloc(128 * 256 * 2);
    _Float16* WcatT = (_Float16*)alloc((size_t)K_TOT * 2048 * 2);
    _Float16* X1    = (_Float16*)alloc((size_t)MT * EH_DIM * 2);
    _Float16* Xemb  = (_Float16*)alloc((size_t)MT * E_DIM * 2);
    float*    cbuf  = (float*)alloc((size_t)S_CELLS * H_DIM * 4);
    _Float16* hraw  = (_Float16*)alloc((size_t)S_CELLS * H_DIM * 2);
    _Float16* hpool = (_Float16*)alloc((size_t)S_CELLS * H_DIM * 2);
    float*    o1    = (float*)alloc((size_t)S_CELLS * 128 * 4);
    float*    o2    = (float*)alloc((size_t)S_CELLS * 256 * 4);
    float*    o3    = (float*)alloc((size_t)S_CELLS * 64 * 4);
    (void)ws_size; (void)in_sizes; (void)n_in; (void)out_size;

    float* dout = (float*)d_out;

    // prep
    k_lane<<<1, 1, 0, stream>>>(lane, Wlg1, blg1, Wlg2, blg2, lc);
    k_wcat<<<(K_TOT * 2048) / 256, 256, 0, stream>>>(Wih, Whh, WcatT);
    k_small_prep<<<(43008 + 255) / 256, 256, 0, stream>>>(bih, bhh, bcat, We1, We1T, We2, We2T);
    k_zero_state<<<(S_CELLS * H_DIM) / 256, 256, 0, stream>>>(cbuf, hpool);

    // embedding MLP (WMMA): [262144,64]->[.,128]->[.,256]
    wgemm_k<1><<<dim3(MT / 128, EH_DIM / 128), 256, 0, stream>>>(
        (const void*)inputData, I_FEAT, We1T, EH_DIM, be1, X1, EH_DIM, I_FEAT, lc);
    wgemm_k<0><<<dim3(MT / 128, E_DIM / 128), 256, 0, stream>>>(
        (const void*)X1, EH_DIM, We2T, E_DIM, be2, Xemb, E_DIM, EH_DIM, lc);

    // recurrent loop
    for (int t = 0; t < T_STEPS; ++t) {
        lstm_step_k<<<dim3(S_CELLS / 128, H_DIM / 64), 256, 0, stream>>>(
            Xemb, hpool, WcatT, bcat, cbuf, hraw, t);
        k_pool<<<(S_CELLS * H_DIM / 8) / 256, 256, 0, stream>>>(hraw, hpool);
    }

    // head
    k_head1<<<(S_CELLS * 128) / 256, 256, 0, stream>>>(hpool, Wout, bout, o1);
    k_mlp<<<(S_CELLS * 256) / 256, 256, 0, stream>>>(o1, 128, Wf1, bf1, o2, 256);
    k_mlp<<<(S_CELLS * 64) / 256, 256, 0, stream>>>(o2, 256, Wf2, bf2, o3, 64);
    k_head4<<<S_CELLS / 256, 256, 0, stream>>>(o3, Wf3, bf3, lc, dout);
    k_final<<<(S_CELLS * H_DIM) / 256, 256, 0, stream>>>(hpool, cbuf, dout);
}
